// BPBookMemory_85109071937682
// MI455X (gfx1250) — compile-verified, matrix operand order checked by reading, not run on hardware
//
#include <hip/hip_runtime.h>
#include <hip/hip_bf16.h>
#include <stdint.h>

#define D_MODEL   512
#define NUM_SLOTS 8192
#define TOPK      8
#define ROWS_PB   16
#define THREADS   256
#define EPS       1e-12f

typedef __attribute__((ext_vector_type(16))) __bf16 v16bf;
typedef __attribute__((ext_vector_type(8)))  float  v8f;

// packed B layout: [tile(512)][kchunk(16)][lane(32)][16 halves]
//   lane<16 : col=lane,    halves 0..7 = K kbase+0..7,  8..15 = K kbase+16..23
//   lane>=16: col=lane-16, halves 0..7 = K kbase+8..15, 8..15 = K kbase+24..31
#define TILE_HALVES (16 * 32 * 16)   // 8192 bf16 per 16-slot tile

// ---------------------------------------------------------------------------
// Kernel 1: normalize memory to bf16 and scatter into WMMA B-fragment order
// ---------------------------------------------------------------------------
__global__ __launch_bounds__(256)
void normalize_memory_pack(const float* __restrict__ mem, __bf16* __restrict__ packB)
{
    const int s = blockIdx.x;
    const int t = threadIdx.x;
    const float v0 = mem[(size_t)s * D_MODEL + t];
    const float v1 = mem[(size_t)s * D_MODEL + 256 + t];
    __shared__ float red[256];
    red[t] = v0 * v0 + v1 * v1;
    __syncthreads();
    for (int off = 128; off > 0; off >>= 1) {
        if (t < off) red[t] += red[t + off];
        __syncthreads();
    }
    const float ninv = 1.0f / fmaxf(sqrtf(red[0]), EPS);

    const int tile = s >> 4;
    const int col  = s & 15;
#pragma unroll
    for (int half = 0; half < 2; ++half) {
        const int   d = t + half * 256;
        const float v = (half ? v1 : v0) * ninv;
        const int chunk = d >> 5;
        const int kin   = d & 31;
        const int g     = kin >> 3;
        const int lane  = col + ((g & 1) << 4);
        const int idx   = (kin & 7) + ((g & 2) ? 8 : 0);
        packB[(((size_t)tile * 16 + chunk) * 32 + lane) * 16 + idx] = (__bf16)v;
    }
}

// ---------------------------------------------------------------------------
// Kernel 2: fused  sim -> top8 -> softmax -> gather -> residual
// block = 16 rows, 8 waves; each wave: 16x64 slots per iter (4 WMMA tiles),
// K-chunk double-buffered so WMMAs overlap next chunk's loads.
// ---------------------------------------------------------------------------
__global__ __launch_bounds__(THREADS)
void retrieve_topk_wmma(const float* __restrict__ x,
                        const float* __restrict__ mem,
                        const __bf16* __restrict__ packB,
                        const float* __restrict__ scale_p,
                        float* __restrict__ out)
{
    __shared__ __align__(16) float  xstage[ROWS_PB][D_MODEL];   // 32 KB raw x
    __shared__ __align__(32) __bf16 aPack[16][32][16];          // 16 KB A frags
    __shared__ __align__(16) float  simbuf[ROWS_PB][512];       // 32 KB
    __shared__ __align__(16) float  candv [THREADS][TOPK];      //  8 KB
    __shared__ __align__(16) int    candi [THREADS][TOPK];      //  8 KB
    __shared__ float redbuf[ROWS_PB][16];
    __shared__ float rinv  [ROWS_PB];
    __shared__ float attw  [ROWS_PB][TOPK];
    __shared__ int   atti  [ROWS_PB][TOPK];

    const int t    = threadIdx.x;
    const int row0 = blockIdx.x * ROWS_PB;

    // ---- stage x tile (16 rows x 512 f32 = 32 KB) via async global->LDS ----
    {
        uint32_t lbase = (uint32_t)(uintptr_t)(&xstage[0][0]);
        const float* gbase = x + (size_t)row0 * D_MODEL;
#pragma unroll
        for (int j = 0; j < 8; ++j) {
            const int c = t + j * THREADS;                 // 16B chunk id
            uint32_t laddr = lbase + c * 16;
            uint64_t ga    = (uint64_t)(uintptr_t)(gbase + c * 4);
            asm volatile("global_load_async_to_lds_b128 %0, %1, off"
                         :: "v"(laddr), "v"(ga) : "memory");
        }
        asm volatile("s_wait_asynccnt 0" ::: "memory");
    }
    __syncthreads();

    // ---- pack x tile into A-fragment order (bf16) + per-row sq norms ----
    {
        const int row   = t >> 4;
        const int chunk = t & 15;
        float ss = 0.f;
#pragma unroll
        for (int kin = 0; kin < 32; ++kin) {
            const float v = xstage[row][chunk * 32 + kin];
            ss += v * v;
            const int g    = kin >> 3;
            const int lane = row + ((g & 1) << 4);
            const int idx  = (kin & 7) + ((g & 2) ? 8 : 0);
            aPack[chunk][lane][idx] = (__bf16)v;
        }
        redbuf[row][chunk] = ss;
    }
    __syncthreads();
    if (t < ROWS_PB) {
        float s = 0.f;
#pragma unroll
        for (int i = 0; i < 16; ++i) s += redbuf[t][i];
        rinv[t] = 1.0f / fmaxf(sqrtf(s), EPS);
    }
    __syncthreads();

    // ---- main loop: 4-tile WMMA blocking + streaming top-8 ----
    const int wave = t >> 5;
    const int lane = t & 31;
    const int half = lane >> 4;
    const int rc   = lane & 15;

    float tv[TOPK]; int ti[TOPK];
#pragma unroll
    for (int k = 0; k < TOPK; ++k) { tv[k] = -__builtin_inff(); ti[k] = 0; }

    for (int it = 0; it < NUM_SLOTS / 512; ++it) {
        const int tbase = it * 32 + wave * 4;   // first of 4 16-slot tiles
        const __bf16* bbase =
            packB + (size_t)tbase * TILE_HALVES + (size_t)lane * 16;

        v8f acc[4];
#pragma unroll
        for (int tl = 0; tl < 4; ++tl) acc[tl] = {};

        // double-buffered fragments: load chunk ch+1 while WMMAing chunk ch
        v16bf aCur, aNxt;
        v16bf bCur[4], bNxt[4];

        aCur = *(const v16bf*)(&aPack[0][lane][0]);
#pragma unroll
        for (int tl = 0; tl < 4; ++tl)
            bCur[tl] = *(const v16bf*)(bbase + (size_t)tl * TILE_HALVES);

#pragma unroll
        for (int ch = 0; ch < 16; ++ch) {
            if (ch + 1 < 16) {
                aNxt = *(const v16bf*)(&aPack[ch + 1][lane][0]);
                const __bf16* bp = bbase + (size_t)(ch + 1) * (32 * 16);
#pragma unroll
                for (int tl = 0; tl < 4; ++tl)
                    bNxt[tl] = *(const v16bf*)(bp + (size_t)tl * TILE_HALVES);
            }
#pragma unroll
            for (int tl = 0; tl < 4; ++tl)
                acc[tl] = __builtin_amdgcn_wmma_f32_16x16x32_bf16(
                              false, aCur, false, bCur[tl], (short)0,
                              acc[tl], false, false);
            aCur = aNxt;
#pragma unroll
            for (int tl = 0; tl < 4; ++tl) bCur[tl] = bNxt[tl];
        }

        __syncthreads();   // previous scan done with simbuf
        {
            const int mbase = half * 8;       // C layout: VGPR r -> row r / r+8
#pragma unroll
            for (int tl = 0; tl < 4; ++tl) {
#pragma unroll
                for (int r = 0; r < 8; ++r) {
                    const int m = mbase + r;
                    simbuf[m][wave * 64 + tl * 16 + rc] = acc[tl][r] * rinv[m];
                }
            }
        }
        __syncthreads();

        // 16 threads per row, 32 columns each: streaming top-8 in registers
        {
            const int srow = t >> 4;
            const int j0   = (t & 15) * 32;
#pragma unroll
            for (int j = 0; j < 32; ++j) {
                const float v = simbuf[srow][j0 + j];
                float mn = tv[0]; int mi = 0;
#pragma unroll
                for (int k = 1; k < TOPK; ++k)
                    if (tv[k] < mn) { mn = tv[k]; mi = k; }
                if (v > mn) { tv[mi] = v; ti[mi] = it * 512 + j0 + j; }
            }
        }
    }

    // ---- merge 16 partial top-8 lists per row, softmax ----
#pragma unroll
    for (int k = 0; k < TOPK; ++k) { candv[t][k] = tv[k]; candi[t][k] = ti[k]; }
    __syncthreads();

    if (t < ROWS_PB) {
        float fv[TOPK]; int fi[TOPK];
#pragma unroll
        for (int k = 0; k < TOPK; ++k) { fv[k] = -__builtin_inff(); fi[k] = 0; }
        for (int s = 0; s < 16; ++s) {
            const int src = t * 16 + s;
#pragma unroll
            for (int k = 0; k < TOPK; ++k) {
                const float v = candv[src][k];
                const int  id = candi[src][k];
                float mn = fv[0]; int mi = 0;
#pragma unroll
                for (int q = 1; q < TOPK; ++q)
                    if (fv[q] < mn) { mn = fv[q]; mi = q; }
                if (v > mn) { fv[mi] = v; fi[mi] = id; }
            }
        }
        float mx = fv[0];
#pragma unroll
        for (int k = 1; k < TOPK; ++k) mx = fmaxf(mx, fv[k]);
        float e[TOPK]; float sum = 0.f;
#pragma unroll
        for (int k = 0; k < TOPK; ++k) { e[k] = __expf(fv[k] - mx); sum += e[k]; }
        const float inv = 1.0f / sum;
#pragma unroll
        for (int k = 0; k < TOPK; ++k) { attw[t][k] = e[k] * inv; atti[t][k] = fi[k]; }
    }
    __syncthreads();

    // ---- gather original fp32 memory rows, weighted sum, residual ----
    {
        const float sc  = scale_p[0];
        const int   row = t >> 4;
        const int   d0  = (t & 15) * 32;
        float w[TOPK]; const float* mrow[TOPK];
#pragma unroll
        for (int k = 0; k < TOPK; ++k) {
            w[k]    = attw[row][k];
            mrow[k] = mem + (size_t)atti[row][k] * D_MODEL + d0;
        }
        float* orow = out + (size_t)(row0 + row) * D_MODEL + d0;
#pragma unroll
        for (int d = 0; d < 32; d += 4) {
            float4 acc = make_float4(0.f, 0.f, 0.f, 0.f);
#pragma unroll
            for (int k = 0; k < TOPK; ++k) {
                const float4 mv = *(const float4*)(mrow[k] + d);
                acc.x += w[k] * mv.x; acc.y += w[k] * mv.y;
                acc.z += w[k] * mv.z; acc.w += w[k] * mv.w;
            }
            const float4 xv = *(const float4*)(&xstage[row][d0 + d]);
            float4 o;
            o.x = xv.x + sc * acc.x; o.y = xv.y + sc * acc.y;
            o.z = xv.z + sc * acc.z; o.w = xv.w + sc * acc.w;
            *(float4*)(orow + d) = o;
        }
    }
}

// ---------------------------------------------------------------------------
extern "C" void kernel_launch(void* const* d_in, const int* in_sizes, int n_in,
                              void* d_out, int out_size, void* d_ws, size_t ws_size,
                              hipStream_t stream)
{
    const float* x   = (const float*)d_in[0];   // [8,4096,512] f32
    const float* mem = (const float*)d_in[1];   // [8192,512]   f32
    const float* sc  = (const float*)d_in[2];   // scalar f32
    float* out = (float*)d_out;                 // [8,4096,512] f32
    __bf16* packB = (__bf16*)d_ws;              // 8192*512*2 = 8 MB scratch

    normalize_memory_pack<<<NUM_SLOTS, 256, 0, stream>>>(mem, packB);

    const int n_rows = in_sizes[0] / D_MODEL;   // 32768
    retrieve_topk_wmma<<<n_rows / ROWS_PB, THREADS, 0, stream>>>(
        x, mem, packB, sc, out);
}
